// ind_KIMIA_75118978007624
// MI455X (gfx1250) — matrix operand order repaired and checked by reading, not compiled, not run on hardware
//
#include <hip/hip_runtime.h>
#include <hip/hip_bf16.h>
#include <math.h>

// ---------------------------------------------------------------------------
// Problem constants (fixed by the reference): B=4096, D=512, KD=448, TR=64,
// NI=16, DK=512. All GEMMs are 4096 x 512 x 512 after padding Wik to 512 cols
// and folding the sin(t) encoding into a per-t bias vector for the K append.
// ---------------------------------------------------------------------------

// gfx1250 async global->LDS copy path (ASYNCcnt), with safe fallback.
#if defined(__HIP_DEVICE_COMPILE__) && __has_builtin(__builtin_amdgcn_global_load_async_to_lds_b128)
#define ATHENA_ASYNC_LDS 1
#else
#define ATHENA_ASYNC_LDS 0
#endif

namespace {
constexpr int Bn  = 4096;
constexpr int Dd  = 512;
constexpr int KDm = 448;
constexpr int TRr = 64;
constexpr int NIi = 16;
constexpr int DKk = 512;

typedef __attribute__((ext_vector_type(16))) __bf16 v16bf;
typedef __attribute__((ext_vector_type(8)))  float  v8f;
typedef int athena_v4i __attribute__((vector_size(16)));   // b128 payload type

// ------------------------- elementwise helpers -----------------------------
__global__ void cvt_f32_bf16_k(const float* __restrict__ s, __bf16* __restrict__ d, int n) {
  int i = blockIdx.x * 256 + threadIdx.x;
  if (i < n) d[i] = (__bf16)s[i];
}

__global__ void pad_wik_k(const float* __restrict__ s, __bf16* __restrict__ d) {
  int i = blockIdx.x * 256 + threadIdx.x;           // over 512*512
  int r = i >> 9, c = i & 511;
  d[i] = (c < KDm) ? (__bf16)s[r * KDm + c] : (__bf16)0.0f;
}

__global__ void fill_ones_k(__bf16* __restrict__ d, int n) {
  int i = blockIdx.x * 256 + threadIdx.x;
  if (i < n) d[i] = (__bf16)1.0f;
}

// bias_t[n] = sum_r sin(t * pi * 2^-r) * Wk[512+r][n]   (t = 0..14)
__global__ void kbias_k(const float* __restrict__ WkF, float* __restrict__ biasK) {
  int t = blockIdx.x, n = threadIdx.x;
  float acc = 0.0f;
  for (int r = 0; r < TRr; ++r) {
    float enc = 3.14159265358979323846f * exp2f(-(float)r);
    acc += sinf((float)t * enc) * WkF[(size_t)(Dd + r) * DKk + n];
  }
  biasK[t * DKk + n] = acc;
}

__device__ __forceinline__ float gelu_tanh(float x) {
  float x3 = x * x * x;
  return 0.5f * x * (1.0f + tanhf(0.7978845608028654f * (x + 0.044715f * x3)));
}

__device__ __forceinline__ void wait_async_zero() {
#if ATHENA_ASYNC_LDS
#if __has_builtin(__builtin_amdgcn_s_wait_asynccnt)
  __builtin_amdgcn_s_wait_asynccnt(0);
#else
  asm volatile("s_wait_asynccnt 0x0" ::: "memory");
#endif
#endif
}

// ------------------------------ WMMA GEMM ----------------------------------
// C[4096 x 512] = act(A[4096 x K] @ W[K x 512] + bias)
// Block tile 128x128, 8 waves (4 in M x 2 in N); each wave owns 32x64 via
// 2x4 v_wmma_f32_16x16x32_bf16 fragments (8 WMMA per k-step). K stepped by
// 32 through double-buffered LDS: A tile staged with async global->LDS b128
// copies (ASYNCcnt), W tile staged transposed [n][k] for contiguous b128
// fragment reads. One barrier per k-step.
constexpr int BM = 128, BN = 128, BK = 32;
constexpr int LDA_S = 40;   // padded LDS strides in halfs (byte stride 80, 16B-aligned)
constexpr int LDB_S = 40;

template <bool HAS_BIAS, bool GELU, bool F32OUT>
__global__ __launch_bounds__(256)
void gemm_bf16_wmma(const __bf16* __restrict__ A, int lda,
                    const __bf16* __restrict__ W, int ldw,
                    void* __restrict__ Cout, int ldc,
                    const float* __restrict__ bias, int Kdim)
{
  __shared__ __bf16 As[2][BM * LDA_S];    // [m][k]
  __shared__ __bf16 Bs[2][BN * LDB_S];    // transposed: [n][k]

  const int tid  = threadIdx.x;
  const int bm   = blockIdx.y * BM;
  const int bn   = blockIdx.x * BN;
  const int wave = tid >> 5;
  const int lane = tid & 31;
  const int wm   = wave & 3;              // 4 waves in M -> 32 rows each
  const int wn   = wave >> 2;             // 2 waves in N -> 64 cols each
  const int lh   = lane & 15;
  const int hi   = lane >> 4;             // lane half-select per ISA layout

  v8f acc[2][4] = {};

  // ---- tile staging (2 A-chunks + 2 W-chunks of 8 halfs per thread) -------
  auto stage = [&](int buf, int k0) {
    #pragma unroll
    for (int i = 0; i < 2; ++i) {
      int c = tid * 2 + i;                       // 512 chunks: 128 rows x 4
      int r = c >> 2, kc = (c & 3) << 3;
      const __bf16* gsrc = A + (size_t)(bm + r) * lda + k0 + kc;
      __bf16*       ldst = &As[buf][r * LDA_S + kc];
#if ATHENA_ASYNC_LDS
      __builtin_amdgcn_global_load_async_to_lds_b128(
          (__attribute__((address_space(1))) athena_v4i*)gsrc,
          (__attribute__((address_space(3))) athena_v4i*)ldst, 0, 0);
#else
      *reinterpret_cast<uint4*>(ldst) = *reinterpret_cast<const uint4*>(gsrc);
#endif
    }
    #pragma unroll
    for (int i = 0; i < 2; ++i) {
      int c = tid * 2 + i;                       // 512 chunks: 32 rows x 16
      int kr = c >> 4, nc = (c & 15) << 3;
      uint4 v = *reinterpret_cast<const uint4*>(W + (size_t)(k0 + kr) * ldw + bn + nc);
      const __bf16* h = reinterpret_cast<const __bf16*>(&v);
      #pragma unroll
      for (int j = 0; j < 8; ++j) Bs[buf][(nc + j) * LDB_S + kr] = h[j];
    }
  };

  stage(0, 0);
  wait_async_zero();
  __syncthreads();

  for (int k0 = 0, it = 0; k0 < Kdim; k0 += BK, ++it) {
    const int cur = it & 1;
    const bool more = (k0 + BK) < Kdim;
    if (more) stage(cur ^ 1, k0 + BK);          // overlap next-tile fill

    // ---- fragments per documented 16-bit A/B VGPR layouts
    union Frag { v16bf v; uint4 q[2]; } a[2], b[4];
    #pragma unroll
    for (int mi = 0; mi < 2; ++mi) {
      int row = wm * 32 + mi * 16 + lh;                    // M = lane&15
      a[mi].q[0] = *reinterpret_cast<const uint4*>(&As[cur][row * LDA_S + (hi ? 8  : 0)]);
      a[mi].q[1] = *reinterpret_cast<const uint4*>(&As[cur][row * LDA_S + (hi ? 24 : 16)]);
    }
    #pragma unroll
    for (int ni = 0; ni < 4; ++ni) {
      int col = wn * 64 + ni * 16 + lh;                    // N = lane&15
      b[ni].q[0] = *reinterpret_cast<const uint4*>(&Bs[cur][col * LDB_S + (hi ? 8  : 0)]);
      b[ni].q[1] = *reinterpret_cast<const uint4*>(&Bs[cur][col * LDB_S + (hi ? 24 : 16)]);
    }
    #pragma unroll
    for (int mi = 0; mi < 2; ++mi)
      #pragma unroll
      for (int ni = 0; ni < 4; ++ni)
        acc[mi][ni] = __builtin_amdgcn_wmma_f32_16x16x32_bf16(
            false, a[mi].v, false, b[ni].v, (short)0, acc[mi][ni], false, false);

    if (more) wait_async_zero();                // own async writes landed
    __syncthreads();                            // publish next buffer
  }

  // ---- epilogue: C layout VGPR r -> M = r + 8*hi, N = lane&15
  float bv[4];
  if constexpr (HAS_BIAS) {
    #pragma unroll
    for (int ni = 0; ni < 4; ++ni)
      bv[ni] = bias[bn + wn * 64 + ni * 16 + lh];
  }
  #pragma unroll
  for (int mi = 0; mi < 2; ++mi)
    #pragma unroll
    for (int ni = 0; ni < 4; ++ni)
      #pragma unroll
      for (int r = 0; r < 8; ++r) {
        int row = bm + wm * 32 + mi * 16 + r + hi * 8;
        int col = bn + wn * 64 + ni * 16 + lh;
        float v = acc[mi][ni][r];
        if constexpr (HAS_BIAS) v += bv[ni];
        if constexpr (GELU)     v = gelu_tanh(v);
        if constexpr (F32OUT)
          ((float*)Cout)[(size_t)row * ldc + col] = v;
        else
          ((__bf16*)Cout)[(size_t)row * ldc + col] = (__bf16)v;
      }
}

// ------------------------------ attention ----------------------------------
// A[b] = softmax(q[b] . K[b,:n] / sqrt(DK)) . V[b,:n]   (n <= 16)
__global__ __launch_bounds__(256)
void attend_bf16(const __bf16* __restrict__ q, const __bf16* __restrict__ Kc,
                 const __bf16* __restrict__ Vc, __bf16* __restrict__ Aout, int n)
{
  const int b    = blockIdx.x;
  const int tid  = threadIdx.x;
  const int wave = tid >> 5, lane = tid & 31;
  __shared__ float s_sc[NIi];
  const float scale = 0.044194173824159216f;   // 1/sqrt(512)

  for (int t = wave; t < n; t += 8) {
    const __bf16* qp = q  + (size_t)b * DKk;
    const __bf16* kp = Kc + ((size_t)b * NIi + t) * DKk;
    float p = 0.0f;
    for (int d = lane; d < DKk; d += 32)
      p += (float)qp[d] * (float)kp[d];
    #pragma unroll
    for (int m = 16; m >= 1; m >>= 1) p += __shfl_xor(p, m, 32);
    if (lane == 0) s_sc[t] = p * scale;
  }
  __syncthreads();

  float pr[NIi];
  float mx = -1e30f;
  for (int t = 0; t < n; ++t) mx = fmaxf(mx, s_sc[t]);
  float sum = 0.0f;
  for (int t = 0; t < n; ++t) { pr[t] = __expf(s_sc[t] - mx); sum += pr[t]; }
  float inv = 1.0f / sum;

  for (int d = tid; d < Dd; d += 256) {
    float acc = 0.0f;
    for (int t = 0; t < n; ++t)
      acc += pr[t] * (float)Vc[((size_t)b * NIi + t) * Dd + d];
    Aout[(size_t)b * Dd + d] = (__bf16)(acc * inv);
  }
}
} // anonymous namespace

// ---------------------------------------------------------------------------
extern "C" void kernel_launch(void* const* d_in, const int* in_sizes, int n_in,
                              void* d_out, int out_size, void* d_ws, size_t ws_size,
                              hipStream_t stream) {
  const float* x    = (const float*)d_in[0];
  const float* Wik  = (const float*)d_in[1];
  const float* Wiv  = (const float*)d_in[2];
  const float* Wq   = (const float*)d_in[3];
  const float* Wk   = (const float*)d_in[4];
  const float* Wv   = (const float*)d_in[5];
  const float* W1   = (const float*)d_in[6];
  const float* W2   = (const float*)d_in[7];
  const float* temb = (const float*)d_in[8];

  // ---- workspace carve-out (all 256B aligned)
  char* p = (char*)d_ws;
  auto carve = [&](size_t bytes) -> char* {
    char* r = p; p += (bytes + 255) & ~(size_t)255; return r;
  };
  __bf16* xb    = (__bf16*)carve((size_t)Bn * Dd * 2);
  __bf16* WikPb = (__bf16*)carve((size_t)Dd * DKk * 2);   // padded to 512 cols
  __bf16* Wivb  = (__bf16*)carve((size_t)Dd * Dd  * 2);
  __bf16* Wqb   = (__bf16*)carve((size_t)Dd * DKk * 2);
  __bf16* Wktb  = (__bf16*)carve((size_t)Dd * DKk * 2);   // top 512 rows of Wk
  __bf16* Wvb   = (__bf16*)carve((size_t)Dd * Dd  * 2);
  __bf16* W1b   = (__bf16*)carve((size_t)Dd * Dd  * 2);
  __bf16* W2b   = (__bf16*)carve((size_t)Dd * Dd  * 2);
  float*  biasK = (float*) carve((size_t)(NIi - 1) * DKk * 4);
  __bf16* qb    = (__bf16*)carve((size_t)Bn * DKk * 2);
  __bf16* Ab    = (__bf16*)carve((size_t)Bn * Dd  * 2);
  __bf16* Hb    = (__bf16*)carve((size_t)Bn * Dd  * 2);
  __bf16* Btb   = (__bf16*)carve((size_t)Bn * Dd  * 2);
  __bf16* Kc    = (__bf16*)carve((size_t)Bn * NIi * DKk * 2);
  __bf16* Vc    = (__bf16*)carve((size_t)Bn * NIi * Dd  * 2);
  (void)ws_size; (void)in_sizes; (void)n_in; (void)out_size;

  const int WW = Dd * DKk;   // 512*512
  auto cvt = [&](const float* s, __bf16* d, int n) {
    cvt_f32_bf16_k<<<(n + 255) / 256, 256, 0, stream>>>(s, d, n);
  };

  // ---- precision staging + constant folding
  cvt(x,   xb,   Bn * Dd);
  cvt(Wiv, Wivb, WW);
  cvt(Wq,  Wqb,  WW);
  cvt(Wk,  Wktb, WW);        // first 512 rows of Wk are contiguous
  cvt(Wv,  Wvb,  WW);
  cvt(W1,  W1b,  WW);
  cvt(W2,  W2b,  WW);
  pad_wik_k<<<WW / 256, 256, 0, stream>>>(Wik, WikPb);
  kbias_k<<<NIi - 1, DKk, 0, stream>>>(Wk, biasK);
  fill_ones_k<<<(Bn * DKk) / 256, 256, 0, stream>>>(qb, Bn * DKk);

  const dim3 ggrid(DKk / BN, Bn / BM);   // (4, 32)
  auto gemm = [&](const __bf16* A, const __bf16* W,
                  __bf16* Cb, int ldc) {               // plain bf16 out
    gemm_bf16_wmma<false, false, false><<<ggrid, 256, 0, stream>>>(
        A, Dd, W, DKk, (void*)Cb, ldc, nullptr, Dd);
  };
  auto gemm_bias = [&](const __bf16* A, const __bf16* W,
                       __bf16* Cb, int ldc, const float* bias) {
    gemm_bf16_wmma<true, false, false><<<ggrid, 256, 0, stream>>>(
        A, Dd, W, DKk, (void*)Cb, ldc, bias, Dd);
  };
  auto gemm_bias_gelu = [&](const __bf16* A, const __bf16* W,
                            __bf16* Cb, int ldc, const float* bias) {
    gemm_bf16_wmma<true, true, false><<<ggrid, 256, 0, stream>>>(
        A, Dd, W, DKk, (void*)Cb, ldc, bias, Dd);
  };
  auto gemm_f32 = [&](const __bf16* A, const __bf16* W,
                      float* Cf, int ldc) {
    gemm_bf16_wmma<false, false, true><<<ggrid, 256, 0, stream>>>(
        A, Dd, W, DKk, (void*)Cf, ldc, nullptr, Dd);
  };

  const int ldKV = NIi * DKk;   // 8192

  // ---- initial cache rows:  K[:,0] = pad(x@Wik),  V[:,0] = x@Wiv
  gemm(xb, WikPb, Kc, ldKV);
  gemm(xb, Wivb,  Vc, ldKV);

  // ---- unrolled recurrence
  for (int t = 0; t < NIi - 1; ++t) {
    attend_bf16<<<Bn, 256, 0, stream>>>(qb, Kc, Vc, Ab, t + 1);
    gemm_bias_gelu(Ab, W1b, Hb, Dd, temb + (size_t)t * Dd);     // gelu(A@W1 + t_emb[t])
    gemm(Hb, W2b, Btb, Dd);                                     // Bt
    gemm_bias(Btb, Wktb, Kc + (size_t)(t + 1) * DKk, ldKV,
              biasK + (size_t)t * DKk);                         // K append (+sin bias)
    gemm(Btb, Wvb, Vc + (size_t)(t + 1) * Dd, ldKV);            // V append
    gemm(Btb, Wqb, qb, DKk);                                    // q update
  }

  // ---- final attend + block -> fp32 output
  attend_bf16<<<Bn, 256, 0, stream>>>(qb, Kc, Vc, Ab, NIi);
  gemm_bias_gelu(Ab, W1b, Hb, Dd, temb + (size_t)NIi * Dd);
  gemm_f32(Hb, W2b, (float*)d_out, Dd);
}
